// GCLSTM_90486370992706
// MI455X (gfx1250) — matrix operand order repaired and checked by reading, not compiled, not run on hardware
//
#include <hip/hip_runtime.h>

typedef __attribute__((ext_vector_type(16))) _Float16 v16h;
typedef __attribute__((ext_vector_type(8)))  _Float16 v8h;
typedef __attribute__((ext_vector_type(8)))  float    v8f;

#define BT_TOT   8192      // B*T
#define CH       12
#define M_BIG    98304     // BT*CH
#define NPAD     208       // 200 padded to 13*16
#define NTILES   13
#define KPAD1    320       // 300 padded to 10*32
#define KPAD2    640       // 600 padded to 20*32
#define LDT      40        // LDS tile stride in halfs (80B: 16B-aligned, conflict-free)

__device__ __forceinline__ float sigf(float x) { return 1.0f / (1.0f + __expf(-x)); }

// ---------------------------------------------------------------------------
// Weight prep: transpose Att1 halves + fuse2 into f16 [n][k] layouts, zero pad.
// ---------------------------------------------------------------------------
__global__ __launch_bounds__(256)
void prep_k(const float* __restrict__ Att1, const float* __restrict__ fuse2,
            _Float16* __restrict__ Btop, _Float16* __restrict__ Bbot,
            _Float16* __restrict__ Bfu) {
  int i = blockIdx.x * 256 + threadIdx.x;
  if (i < NPAD * KPAD1) {
    int n = i / KPAD1, k = i - n * KPAD1;
    bool ok = (n < 200) && (k < 300);
    Btop[i] = ok ? (_Float16)Att1[(size_t)k * 200 + n]         : (_Float16)0.f;
    Bbot[i] = ok ? (_Float16)Att1[(size_t)(300 + k) * 200 + n] : (_Float16)0.f;
  }
  if (i < NPAD * KPAD2) {
    int n = i / KPAD2, k = i - n * KPAD2;
    bool ok = (n < 200) && (k < 600);
    Bfu[i] = ok ? (_Float16)fuse2[(size_t)k * 200 + n] : (_Float16)0.f;
  }
}

// ---------------------------------------------------------------------------
// Target LSTM cell: htarget[bt][0:300] (f32 + padded f16 copy).
// ---------------------------------------------------------------------------
__global__ __launch_bounds__(128)
void lstm_t_k(const float* __restrict__ extras, const float* __restrict__ Wt,
              const float* __restrict__ bti, const float* __restrict__ bth,
              float* __restrict__ htF, _Float16* __restrict__ ht16) {
  int bt = blockIdx.x;
  float s[5];
#pragma unroll
  for (int f = 0; f < 5; f++) s[f] = extras[(size_t)bt * 17 + f];
  for (int d = threadIdx.x; d < KPAD1; d += 128) {
    if (d < 300) {
      float i = bti[d] + bth[d];
      float g = bti[600 + d] + bth[600 + d];
      float o = bti[900 + d] + bth[900 + d];
#pragma unroll
      for (int f = 0; f < 5; f++) {
        i += s[f] * Wt[d * 5 + f];
        g += s[f] * Wt[(600 + d) * 5 + f];
        o += s[f] * Wt[(900 + d) * 5 + f];
      }
      float cc = sigf(i) * tanhf(g);
      float hv = sigf(o) * tanhf(cc);
      htF[(size_t)bt * 300 + d] = hv;
      ht16[(size_t)bt * KPAD1 + d] = (_Float16)hv;
    } else {
      ht16[(size_t)bt * KPAD1 + d] = (_Float16)0.f;
    }
  }
}

// ---------------------------------------------------------------------------
// Per-channel LSTM cell: h[bt][c][0:300] in padded f16. W cached in LDS,
// reused across 16 bt rows per block.
// ---------------------------------------------------------------------------
__global__ __launch_bounds__(256)
void lstm_h_k(const float* __restrict__ li, const float* __restrict__ Wih,
              const float* __restrict__ bih, const float* __restrict__ bhh,
              _Float16* __restrict__ h16) {
  __shared__ float Wl[900 * 8];
  __shared__ float Bl[900];
  __shared__ float Sl[16 * 8];
  int c = blockIdx.y;
  int bt0 = blockIdx.x * 16;
  const float* Wc = Wih + (size_t)c * 1200 * 8;
  for (int idx = threadIdx.x; idx < 7200; idx += 256) {
    int r = idx >> 3, f = idx & 7;
    int g = (r < 300) ? r : r + 300;        // rows: i[0:300], g[600:900], o[900:1200]
    Wl[idx] = Wc[g * 8 + f];
  }
  for (int idx = threadIdx.x; idx < 900; idx += 256) {
    int g = (idx < 300) ? idx : idx + 300;
    Bl[idx] = bih[c * 1200 + g] + bhh[c * 1200 + g];
  }
  for (int idx = threadIdx.x; idx < 128; idx += 256) {
    int btl = idx >> 3, f = idx & 7;
    Sl[idx] = li[((size_t)(bt0 + btl) * 20 + f) * 12 + c];
  }
  __syncthreads();
  for (int j = threadIdx.x; j < 16 * 300; j += 256) {
    int btl = j / 300, d = j - btl * 300;
    const float* s = &Sl[btl * 8];
    float i = Bl[d], g = Bl[300 + d], o = Bl[600 + d];
#pragma unroll
    for (int f = 0; f < 8; f++) {
      i += s[f] * Wl[d * 8 + f];
      g += s[f] * Wl[(300 + d) * 8 + f];
      o += s[f] * Wl[(600 + d) * 8 + f];
    }
    float cc = sigf(i) * tanhf(g);
    float hv = sigf(o) * tanhf(cc);
    h16[((size_t)(bt0 + btl) * CH + c) * KPAD1 + d] = (_Float16)hv;
  }
  for (int j = threadIdx.x; j < 16 * 20; j += 256) {
    int btl = j / 20, k = 300 + (j - btl * 20);
    h16[((size_t)(bt0 + btl) * CH + c) * KPAD1 + k] = (_Float16)0.f;
  }
}

// ---------------------------------------------------------------------------
// WMMA f16 GEMM: C[M][208] = A[M][lda] * Bt[n][k] (+bias). 4 waves/block,
// each wave: 16 rows x 13 N-tiles, K stepped by 32 (v_wmma_f32_16x16x32_f16).
// Staging: CDNA5 async global->LDS copies (ASYNCcnt). A sched_barrier pins all
// 26 ds_load_b128s before the 13 back-to-back WMMAs so waits are staggered.
// ---------------------------------------------------------------------------
union HV16 { v16h v; struct { v8h lo; v8h hi; } s; };

__device__ __forceinline__ void async_g2l_b128(const _Float16* gsrc, const _Float16* ldst) {
  // LDS byte offset = low 32 bits of the generic pointer (flat aperture rule).
  unsigned lds_off = (unsigned)(size_t)ldst;
  asm volatile("global_load_async_to_lds_b128 %0, %1, off"
               :: "v"(lds_off), "v"(gsrc) : "memory");
}

__global__ __launch_bounds__(128)
void wmma_gemm_k(const _Float16* __restrict__ A, int lda,
                 const _Float16* __restrict__ Bt, int ldb,
                 const float* __restrict__ bias,
                 float* __restrict__ C, int ksteps) {
  __shared__ __align__(16) _Float16 tA[64 * LDT];
  __shared__ __align__(16) _Float16 tB[NPAD * LDT];
  const int tid  = threadIdx.x;
  const int wave = tid >> 5;
  const int lane = tid & 31;
  const int m0   = blockIdx.x * 64;
  const int n16  = lane & 15;
  const int half = lane >> 4;

  v8f acc[NTILES];
#pragma unroll
  for (int i = 0; i < NTILES; i++)
#pragma unroll
    for (int j = 0; j < 8; j++) acc[i][j] = 0.f;

  for (int ks = 0; ks < ksteps; ks++) {
    const int kk = ks * 32;
    __syncthreads();                       // LDS safe to overwrite
    // stage A tile: 64 rows x 32 halfs (64B/row) as async 16B copies
#pragma unroll
    for (int idx = tid; idx < 64 * 4; idx += 128) {
      int r = idx >> 2, p = idx & 3;
      async_g2l_b128(&A[(size_t)(m0 + r) * lda + kk + p * 8], &tA[r * LDT + p * 8]);
    }
    // stage B tile: 208 rows(n) x 32 halfs
    for (int idx = tid; idx < NPAD * 4; idx += 128) {
      int n = idx >> 2, p = idx & 3;
      async_g2l_b128(&Bt[(size_t)n * ldb + kk + p * 8], &tB[n * LDT + p * 8]);
    }
    if (ks + 1 < ksteps) {   // global_prefetch next A K-slice
      __builtin_prefetch(&A[(size_t)(m0 + (tid >> 1)) * lda + kk + 32 + (tid & 1) * 16], 0, 0);
    }
    asm volatile("s_wait_asynccnt 0x0" ::: "memory");  // this wave's copies done
    __syncthreads();                                   // all waves' copies done

    // A fragment: lane-half holds K blocks {8h..8h+7} and {16+8h..16+8h+7}
    const int mrow = (wave << 4) + n16;
    HV16 a;
    a.s.lo = *(const v8h*)&tA[mrow * LDT + half * 8];
    a.s.hi = *(const v8h*)&tA[mrow * LDT + 16 + half * 8];

    // Issue ALL fragment loads (26 ds_load_b128, in-order returns) ...
    HV16 bf[NTILES];
#pragma unroll
    for (int nt = 0; nt < NTILES; nt++) {
      const _Float16* bp = &tB[(nt * 16 + n16) * LDT + half * 8];
      bf[nt].s.lo = *(const v8h*)bp;
      bf[nt].s.hi = *(const v8h*)(bp + 16);
    }
#if __has_builtin(__builtin_amdgcn_sched_barrier)
    __builtin_amdgcn_sched_barrier(0);     // ... and pin them above the WMMAs
#endif
#pragma unroll
    for (int nt = 0; nt < NTILES; nt++) {
      acc[nt] = __builtin_amdgcn_wmma_f32_16x16x32_f16(
          false, a.v, false, bf[nt].v, (short)0, acc[nt], false, false);
    }
  }

  // epilogue: D vgpr r -> row m0 + wave*16 + r + 8*half, col nt*16 + n16
#pragma unroll
  for (int nt = 0; nt < NTILES; nt++) {
    int col = nt * 16 + n16;
    float bv = (bias != nullptr && col < 200) ? bias[col] : 0.f;
#pragma unroll
    for (int r = 0; r < 8; r++) {
      int row = m0 + (wave << 4) + r + half * 8;
      C[(size_t)row * NPAD + col] = acc[nt][r] + bv;
    }
  }
}

// ---------------------------------------------------------------------------
// score[m] = relu( sum_n relu(Ga[bt][n]+Gb[m][n]) * Att2[n] + wij.Att2[200:202] + ba2 )
// (ba1 already folded into Ga)
// ---------------------------------------------------------------------------
__global__ __launch_bounds__(256)
void score_k(const float* __restrict__ Ga, const float* __restrict__ Gb,
             const float* __restrict__ li, const float* __restrict__ Att2,
             const float* __restrict__ ba2, float* __restrict__ score) {
  int m = blockIdx.x * 256 + threadIdx.x;
  if (m >= M_BIG) return;
  int bt = m / 12, c = m - bt * 12;
  float w0 = li[((size_t)bt * 20 + 7) * 12 + c];   // fc[200]
  float w1 = li[((size_t)bt * 20 + 6) * 12 + c];   // fc[201]
  float acc = ba2[0] + w0 * Att2[200] + w1 * Att2[201];
  const float* ga = Ga + (size_t)bt * NPAD;
  const float* gb = Gb + (size_t)m * NPAD;
  for (int n = 0; n < 200; n++) {
    float v = ga[n] + gb[n];
    v = v > 0.f ? v : 0.f;
    acc += v * Att2[n];
  }
  score[m] = acc > 0.f ? acc : 0.f;
}

// ---------------------------------------------------------------------------
// softmax over the 12 channels per (t,b); stored [t][c*128+b] so the
// reference's reshape-scramble becomes wlin[t*1536 + b*12 + c] on read.
// ---------------------------------------------------------------------------
__global__ __launch_bounds__(256)
void softmax_k(const float* __restrict__ score, float* __restrict__ wlin) {
  int tb = blockIdx.x * 256 + threadIdx.x;
  if (tb >= BT_TOT) return;
  int t = tb >> 7, b = tb & 127;
  float s[12]; float mx = -1e30f;
#pragma unroll
  for (int c = 0; c < 12; c++) {
    s[c] = score[((size_t)(b * 64 + t)) * 12 + c];
    mx = fmaxf(mx, s[c]);
  }
  float sum = 0.f;
#pragma unroll
  for (int c = 0; c < 12; c++) { s[c] = __expf(s[c] - mx); sum += s[c]; }
  float inv = 1.f / sum;
#pragma unroll
  for (int c = 0; c < 12; c++) wlin[(size_t)t * 1536 + c * 128 + b] = s[c] * inv;
}

// ---------------------------------------------------------------------------
// fusion[bt][d] = sum_c h[bt][c][d]*w_scrambled ; fcat16 = [fusion | htarget | 0pad]
// ---------------------------------------------------------------------------
__global__ __launch_bounds__(128)
void fusion_k(const _Float16* __restrict__ h16, const float* __restrict__ htF,
              const float* __restrict__ wlin, _Float16* __restrict__ fcat) {
  int bt = blockIdx.x;
  int b = bt >> 6, t = bt & 63;
  __shared__ float w[12];
  if (threadIdx.x < 12) w[threadIdx.x] = wlin[(size_t)t * 1536 + b * 12 + threadIdx.x];
  __syncthreads();
  for (int d = threadIdx.x; d < 300; d += 128) {
    float acc = 0.f;
#pragma unroll
    for (int c = 0; c < 12; c++)
      acc += (float)h16[((size_t)bt * CH + c) * KPAD1 + d] * w[c];
    fcat[(size_t)bt * KPAD2 + d]       = (_Float16)acc;
    fcat[(size_t)bt * KPAD2 + 300 + d] = (_Float16)htF[(size_t)bt * 300 + d];
  }
  for (int d = threadIdx.x; d < 40; d += 128)
    fcat[(size_t)bt * KPAD2 + 600 + d] = (_Float16)0.f;
}

// ---------------------------------------------------------------------------
// predict + transposed output write (+labels passthrough)
// ---------------------------------------------------------------------------
__global__ __launch_bounds__(256)
void predict_k(const float* __restrict__ Fo, const float* __restrict__ Wout,
               const float* __restrict__ biasout, const float* __restrict__ labels,
               float* __restrict__ out) {
  int bt = blockIdx.x * 256 + threadIdx.x;
  if (bt >= BT_TOT) return;
  float acc = biasout[0];
  const float* f = Fo + (size_t)bt * NPAD;
  for (int n = 0; n < 200; n++) acc += f[n] * Wout[n];
  int b = bt >> 6, t = bt & 63;
  out[(size_t)t * 128 + b] = acc;                    // predicts (T,B,1)
  out[BT_TOT + (size_t)t * 128 + b] = labels[bt];    // labels_out (T,B,1)
}

// ---------------------------------------------------------------------------
extern "C" void kernel_launch(void* const* d_in, const int* in_sizes, int n_in,
                              void* d_out, int out_size, void* d_ws, size_t ws_size,
                              hipStream_t stream) {
  const float* li      = (const float*)d_in[0];
  const float* labels  = (const float*)d_in[1];
  const float* extras  = (const float*)d_in[2];
  // d_in[3]=DisM, d_in[4]=AngleM: unused by the reference
  const float* W_ih    = (const float*)d_in[5];
  const float* b_ih    = (const float*)d_in[6];
  const float* b_hh    = (const float*)d_in[7];
  const float* Wt_ih   = (const float*)d_in[8];
  const float* bt_ih   = (const float*)d_in[9];
  const float* bt_hh   = (const float*)d_in[10];
  const float* Att1    = (const float*)d_in[11];
  const float* ba1     = (const float*)d_in[12];
  const float* Att2    = (const float*)d_in[13];
  const float* ba2     = (const float*)d_in[14];
  const float* fuse2   = (const float*)d_in[15];
  const float* biasf2  = (const float*)d_in[16];
  const float* Wout    = (const float*)d_in[17];
  const float* biasout = (const float*)d_in[18];
  float* out = (float*)d_out;

  char* p = (char*)d_ws;
  auto carve = [&](size_t n) -> char* {
    char* r = p; p += (n + 255) & ~(size_t)255; return r;
  };
  float*    htF   = (float*)   carve((size_t)BT_TOT * 300 * 4);
  _Float16* ht16  = (_Float16*)carve((size_t)BT_TOT * KPAD1 * 2);
  _Float16* h16   = (_Float16*)carve((size_t)M_BIG * KPAD1 * 2);
  _Float16* Btop  = (_Float16*)carve((size_t)NPAD * KPAD1 * 2);
  _Float16* Bbot  = (_Float16*)carve((size_t)NPAD * KPAD1 * 2);
  _Float16* Bfu   = (_Float16*)carve((size_t)NPAD * KPAD2 * 2);
  float*    Ga    = (float*)   carve((size_t)BT_TOT * NPAD * 4);
  float*    Gb    = (float*)   carve((size_t)M_BIG * NPAD * 4);
  float*    sc    = (float*)   carve((size_t)M_BIG * 4);
  float*    wlin  = (float*)   carve((size_t)64 * 1536 * 4);
  _Float16* fcat  = (_Float16*)carve((size_t)BT_TOT * KPAD2 * 2);
  float*    Fo    = (float*)   carve((size_t)BT_TOT * NPAD * 4);

  prep_k  <<<(NPAD * KPAD2 + 255) / 256, 256, 0, stream>>>(Att1, fuse2, Btop, Bbot, Bfu);
  lstm_t_k<<<BT_TOT, 128, 0, stream>>>(extras, Wt_ih, bt_ih, bt_hh, htF, ht16);
  lstm_h_k<<<dim3(BT_TOT / 16, 12), 256, 0, stream>>>(li, W_ih, b_ih, b_hh, h16);
  // Ga = htarget . Att1_top + ba1   (M=8192)
  wmma_gemm_k<<<BT_TOT / 64, 128, 0, stream>>>(ht16, KPAD1, Btop, KPAD1, ba1, Ga, KPAD1 / 32);
  // Gb = h . Att1_bot               (M=98304)
  wmma_gemm_k<<<M_BIG / 64, 128, 0, stream>>>(h16, KPAD1, Bbot, KPAD1, nullptr, Gb, KPAD1 / 32);
  score_k  <<<M_BIG / 256, 256, 0, stream>>>(Ga, Gb, li, Att2, ba2, sc);
  softmax_k<<<BT_TOT / 256, 256, 0, stream>>>(sc, wlin);
  fusion_k <<<BT_TOT, 128, 0, stream>>>(h16, htF, wlin, fcat);
  // Fo = fcat . fuse2 + biasf2      (M=8192, K=640)
  wmma_gemm_k<<<BT_TOT / 64, 128, 0, stream>>>(fcat, KPAD2, Bfu, KPAD2, biasf2, Fo, KPAD2 / 32);
  predict_k<<<BT_TOT / 256, 256, 0, stream>>>(Fo, Wout, biasout, labels, out);
}
